// JPLModule_34660386079161
// MI455X (gfx1250) — matrix operand order repaired and checked by reading, not compiled, not run on hardware
//
#include <hip/hip_runtime.h>
#include <hip/hip_bf16.h>
#include <math.h>

// Problem constants
#define CC    1000
#define UU    10
#define DD    512
#define BB    4096
#define PP    (CC * UU)          // 10000
#define ALPHA 32.0f
#define DELTA 0.1f
#define SHIFT 35.2f              // ALPHA * (1 + DELTA), upper bound of z_pos/z_neg

// Tiling: 128 batch rows x 80 proxy cols (8 classes) per block, 8 waves,
// each wave owns 16 rows x 80 cols = 5 WMMA accumulators, K-step 32.
#define ROWS_PER_BLK 128
#define COLS_PER_BLK 80          // 8 classes * 10 proxies = 5 x 16
#define NT 5                     // column tiles of 16 per wave

typedef _Float16 v16h __attribute__((ext_vector_type(16)));
typedef _Float16 v8h  __attribute__((ext_vector_type(8)));
typedef float    v8f  __attribute__((ext_vector_type(8)));

__device__ __forceinline__ float softplus_f(float x) {
    return x > 20.0f ? x : log1pf(expf(x));
}

// ---------------------------------------------------------------- utilities
__global__ __launch_bounds__(256) void zero_kernel(float* pos_acc, float* neg_acc, int* ccount) {
    int i = blockIdx.x * 256 + threadIdx.x;
    if (i < PP) { pos_acc[i] = 0.0f; neg_acc[i] = 0.0f; }
    if (i < CC) ccount[i] = 0;
}

__global__ __launch_bounds__(256) void count_kernel(const int* __restrict__ labels, int* ccount) {
    int i = blockIdx.x * 256 + threadIdx.x;
    if (i < BB) atomicAdd(&ccount[labels[i]], 1);
}

// Row-normalize (L2, clipped at 1e-12) and convert fp32 -> fp16.
// One block per row of 512; 256 threads, 2 elements each.
__global__ __launch_bounds__(256) void normalize_rows_kernel(const float* __restrict__ in,
                                                             _Float16* __restrict__ out) {
    const int row = blockIdx.x;
    const float* r = in + (size_t)row * DD;
    const int t = threadIdx.x;
    float a = r[t];
    float b = r[t + 256];
    float ss = a * a + b * b;
#pragma unroll
    for (int off = 16; off > 0; off >>= 1) ss += __shfl_down(ss, off, 32);
    __shared__ float part[8];
    __shared__ float inv_s;
    if ((t & 31) == 0) part[t >> 5] = ss;
    __syncthreads();
    if (t == 0) {
        float tot = 0.0f;
#pragma unroll
        for (int i = 0; i < 8; ++i) tot += part[i];
        inv_s = 1.0f / fmaxf(sqrtf(tot), 1e-12f);
    }
    __syncthreads();
    const float inv = inv_s;
    out[(size_t)row * DD + t]       = (_Float16)(a * inv);
    out[(size_t)row * DD + t + 256] = (_Float16)(b * inv);
}

// --------------------------------------------- fused GEMM + logits + loss partials
__global__ __launch_bounds__(256) void fused_gemm_kernel(const _Float16* __restrict__ xh,
                                                         const _Float16* __restrict__ ph,
                                                         const int* __restrict__ labels,
                                                         float* __restrict__ logits,
                                                         float* __restrict__ pos_acc,
                                                         float* __restrict__ neg_acc) {
    __shared__ float simLds[8][16][COLS_PER_BLK];   // 40 KB
    __shared__ int   labLds[ROWS_PER_BLK];

    const int tid  = threadIdx.x;
    const int wave = tid >> 5;
    const int lane = tid & 31;
    const int lh   = lane >> 4;          // half-wave selector (0/1)
    const int lm   = lane & 15;

    const int Rblk = blockIdx.y * ROWS_PER_BLK;
    const int Cblk = blockIdx.x * COLS_PER_BLK;

    if (tid < ROWS_PER_BLK) labLds[tid] = labels[Rblk + tid];

    // ---- WMMA GEMM: this wave computes rows [R, R+16) x cols [Cblk, Cblk+80)
    const int R = Rblk + wave * 16;
    const v8f vzero = {0.f, 0.f, 0.f, 0.f, 0.f, 0.f, 0.f, 0.f};
    v8f acc[NT];
#pragma unroll
    for (int t = 0; t < NT; ++t) acc[t] = vzero;

    // A operand base: lane holds row (R + lm); 16 contiguous K halfs at lh*16
    const _Float16* aRow = xh + (size_t)(R + lm) * DD + lh * 16;

    for (int kb = 0; kb < DD; kb += 32) {
        union { v16h v; v8h h[2]; } A;
        A.h[0] = *(const v8h*)(aRow + kb);
        A.h[1] = *(const v8h*)(aRow + kb + 8);
#pragma unroll
        for (int t = 0; t < NT; ++t) {
            // B[k,n] = p[n,k]: lane holds column (Cblk + t*16 + lm), 16 contiguous K halfs
            const v16h Bv = *(const v16h*)(ph + (size_t)(Cblk + t * 16 + lm) * DD + kb + lh * 16);
            acc[t] = __builtin_amdgcn_wmma_f32_16x16x32_f16(
                false, A.v, false, Bv, (short)0, acc[t], false, false);
        }
    }

    // ---- dump D fragments to LDS: acc[t][r] = sim(R + r + 8*lh, Cblk + t*16 + lm)
#pragma unroll
    for (int t = 0; t < NT; ++t)
#pragma unroll
        for (int r = 0; r < 8; ++r)
            simLds[wave][r + 8 * lh][t * 16 + lm] = acc[t][r];
    __syncthreads();

    // ---- logits: softmax-weighted aggregation over U=10 proxies per class
    // 128 rows x 8 classes = 1024 outputs; 4 per thread.
#pragma unroll
    for (int it = 0; it < 4; ++it) {
        const int idx = it * 256 + tid;
        const int rl  = idx >> 3;            // local row 0..127
        const int cl  = idx & 7;             // local class 0..7
        const float* srow = &simLds[rl >> 4][rl & 15][cl * UU];
        float m = srow[0];
#pragma unroll
        for (int u = 1; u < UU; ++u) m = fmaxf(m, srow[u]);
        float se = 0.0f, wsum = 0.0f;
#pragma unroll
        for (int u = 0; u < UU; ++u) {
            float e = expf(srow[u] - m);
            se += e;
            wsum += e * srow[u];
        }
        logits[(size_t)(Rblk + rl) * CC + blockIdx.x * 8 + cl] = wsum / se;
    }

    // ---- loss partials: fixed-shift sum-exp per proxy column, 2 atomics each
    if (tid < COLS_PER_BLK) {
        const int cls = blockIdx.x * 8 + tid / UU;
        float ps = 0.0f, ns = 0.0f;
        for (int rl = 0; rl < ROWS_PER_BLK; ++rl) {
            const float s = simLds[rl >> 4][rl & 15][tid];
            if (labLds[rl] == cls) ps += expf(-ALPHA * (s - DELTA) - SHIFT);
            else                   ns += expf( ALPHA * (s + DELTA) - SHIFT);
        }
        atomicAdd(&pos_acc[Cblk + tid], ps);
        atomicAdd(&neg_acc[Cblk + tid], ns);
    }
}

// ---------------------------------------------------------------- finalize loss
__global__ __launch_bounds__(256) void finalize_kernel(const float* __restrict__ pos_acc,
                                                       const float* __restrict__ neg_acc,
                                                       const int* __restrict__ ccount,
                                                       float* __restrict__ out_loss) {
    const int t = threadIdx.x;
    float psum = 0.0f, nsum = 0.0f, hcnt = 0.0f;
    for (int j = t; j < PP; j += 256) {
        const bool hp = ccount[j / UU] > 0;
        if (hp) {
            hcnt += 1.0f;
            const float sp = pos_acc[j];
            if (sp > 0.0f) psum += softplus_f(SHIFT + logf(sp));
        }
        const float sn = neg_acc[j];
        if (sn > 0.0f) nsum += softplus_f(SHIFT + logf(sn));
    }
#pragma unroll
    for (int off = 16; off > 0; off >>= 1) {
        psum += __shfl_down(psum, off, 32);
        nsum += __shfl_down(nsum, off, 32);
        hcnt += __shfl_down(hcnt, off, 32);
    }
    __shared__ float red[8][3];
    if ((t & 31) == 0) { red[t >> 5][0] = psum; red[t >> 5][1] = nsum; red[t >> 5][2] = hcnt; }
    __syncthreads();
    if (t == 0) {
        float p = 0.0f, n = 0.0f, h = 0.0f;
#pragma unroll
        for (int i = 0; i < 8; ++i) { p += red[i][0]; n += red[i][1]; h += red[i][2]; }
        out_loss[0] = p / fmaxf(h, 1.0f) + n * (1.0f / (float)PP);
    }
}

// ---------------------------------------------------------------- launch
extern "C" void kernel_launch(void* const* d_in, const int* in_sizes, int n_in,
                              void* d_out, int out_size, void* d_ws, size_t ws_size,
                              hipStream_t stream) {
    const float* features = (const float*)d_in[0];   // [4096, 512] f32
    const int*   labels   = (const int*)d_in[1];     // [4096] i32
    const float* proxies  = (const float*)d_in[2];   // [10000, 512] f32
    float* out = (float*)d_out;                      // logits [4096*1000] + loss [1]

    // Workspace layout (~14.3 MB)
    _Float16* xh      = (_Float16*)d_ws;                       // 4096*512 f16
    _Float16* ph      = xh + (size_t)BB * DD;                  // 10000*512 f16
    float*    pos_acc = (float*)(ph + (size_t)PP * DD);        // 10000 f32
    float*    neg_acc = pos_acc + PP;                          // 10000 f32
    int*      ccount  = (int*)(neg_acc + PP);                  // 1000 i32

    hipLaunchKernelGGL(zero_kernel, dim3((PP + 255) / 256), dim3(256), 0, stream,
                       pos_acc, neg_acc, ccount);
    hipLaunchKernelGGL(count_kernel, dim3(BB / 256), dim3(256), 0, stream, labels, ccount);
    hipLaunchKernelGGL(normalize_rows_kernel, dim3(BB), dim3(256), 0, stream, features, xh);
    hipLaunchKernelGGL(normalize_rows_kernel, dim3(PP), dim3(256), 0, stream, proxies, ph);
    hipLaunchKernelGGL(fused_gemm_kernel,
                       dim3(PP / COLS_PER_BLK, BB / ROWS_PER_BLK), dim3(256), 0, stream,
                       xh, ph, labels, out, pos_acc, neg_acc);
    hipLaunchKernelGGL(finalize_kernel, dim3(1), dim3(256), 0, stream,
                       pos_acc, neg_acc, ccount, out + (size_t)BB * CC);
}
